// GPT_22273700397400
// MI455X (gfx1250) — matrix operand order repaired and checked by reading, not compiled, not run on hardware
//
#include <hip/hip_runtime.h>
#include <hip/hip_bf16.h>
#include <math.h>

// ---------------- model constants ----------------
#define BB   4
#define SS   1024
#define DD   768
#define NH   12
#define HDIM 64
#define NL   12
#define VV   50257
#define FF   3072
#define MTOT (BB * SS)   // 4096 rows of activations (always multiple of 128)

// ---------------- vector types (trivial, union-safe) ----------------
typedef __attribute__((ext_vector_type(16))) __bf16 bf16x16;
typedef __attribute__((ext_vector_type(8)))  float  v8f;
typedef __attribute__((ext_vector_type(4)))  int    i4v;
typedef __attribute__((ext_vector_type(4)))  float  f4v;

union FragU { bf16x16 v; i4v q[2]; };

__device__ __forceinline__ unsigned short f2bf(float f) {
  union { float f; unsigned u; } x; x.f = f;
  unsigned r = x.u + 0x7FFFu + ((x.u >> 16) & 1u);   // round-to-nearest-even
  return (unsigned short)(r >> 16);
}
__device__ __forceinline__ int pack2(float lo, float hi) {
  return (int)(((unsigned)f2bf(hi) << 16) | (unsigned)f2bf(lo));
}

// A-matrix fragment (16x32 bf16): lane holds k = kbase..kbase+7 and kbase+16..kbase+23
__device__ __forceinline__ bf16x16 load_a_frag(const unsigned short* rowp, int kbase) {
  FragU f;
  f.q[0] = *(const i4v*)(rowp + kbase);
  f.q[1] = *(const i4v*)(rowp + kbase + 16);
  return f.v;
}
// B-matrix fragment (32x16 bf16, LDS stored n-major): lane holds 16 consecutive k
__device__ __forceinline__ bf16x16 load_b_frag(const unsigned short* rowp, int k0) {
  FragU f;
  f.q[0] = *(const i4v*)(rowp + k0);
  f.q[1] = *(const i4v*)(rowp + k0 + 8);
  return f.v;
}

// ---------------- embedding: h = wte[x] + wpe[pos] ----------------
__global__ __launch_bounds__(256) void embed_kernel(
    const int* __restrict__ x, const float* __restrict__ wte,
    const float* __restrict__ wpe, float* __restrict__ h)
{
  int row = blockIdx.x;             // b*SS + s
  int s   = row & (SS - 1);
  int tok = x[row];
  const float* te = wte + (size_t)tok * DD;
  const float* pe = wpe + (size_t)s   * DD;
  float* hr = h + (size_t)row * DD;
#pragma unroll
  for (int i = 0; i < 3; ++i) {
    int d = threadIdx.x + i * 256;
    hr[d] = te[d] + pe[d];
  }
}

// ---------------- LayerNorm over D=768, one block per row ----------------
__global__ __launch_bounds__(256) void ln_kernel(
    const float* __restrict__ x, const float* __restrict__ w,
    const float* __restrict__ b, float* __restrict__ y)
{
  __shared__ float sbuf[256];
  int row = blockIdx.x;
  int tid = threadIdx.x;
  const float* xr = x + (size_t)row * DD;
  float v0 = xr[tid], v1 = xr[tid + 256], v2 = xr[tid + 512];

  sbuf[tid] = v0 + v1 + v2;
  __syncthreads();
  for (int off = 128; off > 0; off >>= 1) {
    if (tid < off) sbuf[tid] += sbuf[tid + off];
    __syncthreads();
  }
  float mean = sbuf[0] * (1.0f / DD);
  __syncthreads();

  float d0 = v0 - mean, d1 = v1 - mean, d2 = v2 - mean;
  sbuf[tid] = d0 * d0 + d1 * d1 + d2 * d2;
  __syncthreads();
  for (int off = 128; off > 0; off >>= 1) {
    if (tid < off) sbuf[tid] += sbuf[tid + off];
    __syncthreads();
  }
  float rstd = rsqrtf(sbuf[0] * (1.0f / DD) + 1e-5f);

  float* yr = y + (size_t)row * DD;
  yr[tid]       = d0 * rstd * w[tid]       + b[tid];
  yr[tid + 256] = d1 * rstd * w[tid + 256] + b[tid + 256];
  yr[tid + 512] = d2 * rstd * w[tid + 512] + b[tid + 512];
}

// ---------------- WMMA GEMM: C = act(A[MxK] @ W[KxN] + bias) (+res) ----------------
// REQUIRES: M % 128 == 0, K % 32 == 0 (true for all launches here).
// wmode 0: W row-major [K,N].  wmode 1: W head-major [H, K, 64] (QKV weights).
// Block tile 128x128, 8 waves = 4(M) x 2(N); wave tile 32x64 = 8 WMMA / K-step.
__global__ __launch_bounds__(256) void gemm_wmma(
    const float* __restrict__ A, const float* __restrict__ W,
    const float* __restrict__ bias, const float* __restrict__ res,
    float* __restrict__ C, int M, int N, int K, int wmode, int dogelu)
{
  __shared__ unsigned short As[128][32];   // 8 KB   (row-major, k contiguous)
  __shared__ unsigned short Bs[128][32];   // 8 KB   (n-major, k contiguous)

  const int tid  = threadIdx.x;
  const int wave = tid >> 5, lane = tid & 31;
  const int hl   = lane >> 4, ln = lane & 15;
  const int wm   = wave & 3,  wn = wave >> 2;       // 4 waves in M, 2 in N
  const int m0   = blockIdx.y * 128;
  const int n0   = blockIdx.x * 128;

  // staging coordinates (computed once)
  const int srow = tid >> 1;                 // 0..127 (A row, and Bs n)
  const int skh  = (tid & 1) << 4;           // 0 or 16 (k-half)
  const int gn   = n0 + srow;
  const size_t wstride = (wmode == 0) ? (size_t)N : (size_t)HDIM;
  const float* wp = (wmode == 0)
      ? (W + gn)
      : (W + ((size_t)(gn >> 6) * K) * HDIM + (gn & 63));
  const float* ap = A + (size_t)(m0 + srow) * K + skh;

  v8f acc[2][4] = {};

  for (int k0 = 0; k0 < K; k0 += 32) {
    // ---- stage A tile: 16 contiguous f32 -> 4x b128 loads, 2x b128 LDS stores ----
    {
      const float* src = ap + k0;
      f4v a0 = *(const f4v*)(src + 0);
      f4v a1 = *(const f4v*)(src + 4);
      f4v a2 = *(const f4v*)(src + 8);
      f4v a3 = *(const f4v*)(src + 12);
      i4v p0 = { pack2(a0.x, a0.y), pack2(a0.z, a0.w),
                 pack2(a1.x, a1.y), pack2(a1.z, a1.w) };
      i4v p1 = { pack2(a2.x, a2.y), pack2(a2.z, a2.w),
                 pack2(a3.x, a3.y), pack2(a3.z, a3.w) };
      *(i4v*)&As[srow][skh]     = p0;
      *(i4v*)&As[srow][skh + 8] = p1;
    }
    // ---- stage B tile transposed (n-major): 16 strided f32, packed stores ----
    if (gn < N) {
      float t[16];
#pragma unroll
      for (int j = 0; j < 16; ++j) t[j] = wp[(size_t)(k0 + skh + j) * wstride];
      i4v p0 = { pack2(t[0],  t[1]),  pack2(t[2],  t[3]),
                 pack2(t[4],  t[5]),  pack2(t[6],  t[7]) };
      i4v p1 = { pack2(t[8],  t[9]),  pack2(t[10], t[11]),
                 pack2(t[12], t[13]), pack2(t[14], t[15]) };
      *(i4v*)&Bs[srow][skh]     = p0;
      *(i4v*)&Bs[srow][skh + 8] = p1;
    } else {
      i4v z = { 0, 0, 0, 0 };
      *(i4v*)&Bs[srow][skh]     = z;
      *(i4v*)&Bs[srow][skh + 8] = z;
    }
    // prefetch next A K-slab (emits global_prefetch_b8)
    if (k0 + 32 < K) __builtin_prefetch(ap + k0 + 32, 0, 0);
    __syncthreads();

    // ---- compute: 2 A-frags x 4 B-frags = 8 WMMA ----
    bf16x16 afrag[2];
#pragma unroll
    for (int mi = 0; mi < 2; ++mi)
      afrag[mi] = load_a_frag(&As[wm * 32 + mi * 16 + ln][0], hl * 8);
#pragma unroll
    for (int t = 0; t < 4; ++t) {
      bf16x16 bfrag = load_a_frag(&Bs[wn * 64 + t * 16 + ln][0], hl * 8);
#pragma unroll
      for (int mi = 0; mi < 2; ++mi)
        acc[mi][t] = __builtin_amdgcn_wmma_f32_16x16x32_bf16(
            false, afrag[mi], false, bfrag, (short)0, acc[mi][t], false, false);
    }
    __syncthreads();
  }

  // ---- epilogue: bias (+gelu) (+residual); M-guard not needed ----
#pragma unroll
  for (int t = 0; t < 4; ++t) {
    int col = n0 + wn * 64 + t * 16 + ln;
    if (col >= N) continue;
    float bv = bias[col];
#pragma unroll
    for (int mi = 0; mi < 2; ++mi) {
#pragma unroll
      for (int r = 0; r < 8; ++r) {
        int row = m0 + wm * 32 + mi * 16 + hl * 8 + r;
        float v = acc[mi][t][r] + bv;
        if (dogelu) v = 0.5f * v * (1.0f + erff(v * 0.70710678118f));
        size_t idx = (size_t)row * N + col;
        if (res) v += res[idx];
        C[idx] = v;
      }
    }
  }
}

// ---------------- fused flash attention (causal), WMMA both matmuls ----------------
// q,k,v,o layout: [B, S, H, 64] (== [B,S,D] with heads concatenated)
// scale = 1/sqrt(D) = 1/sqrt(768) folded into Q (reference divides by sqrt(dim)).
__global__ __launch_bounds__(256) void attn_kernel(
    const float* __restrict__ q, const float* __restrict__ k,
    const float* __restrict__ v, float* __restrict__ o)
{
  __shared__ unsigned short Qs[128][64];      // 16 KB  [qrow][e]
  __shared__ unsigned short Ks[32][64];       //  4 KB  [key][e]   (B-frag layout for QK^T)
  __shared__ unsigned short Vs[64][32];       //  4 KB  [e][key]   (B-frag layout for P*V)
  __shared__ unsigned short Ps[8][16][32];    //  8 KB  per-wave P tile (A-frag layout)

  const int tid  = threadIdx.x;
  const int wave = tid >> 5, lane = tid & 31;
  const int hl   = lane >> 4, ln = lane & 15;
  const int b    = blockIdx.z, hh = blockIdx.y;
  const int q0   = blockIdx.x * 128;
  const float rscale = 0.03608439182f;        // 1/sqrt(768)

  // ---- stage Q (scaled): 32 f32 per thread -> 4x b128 LDS stores ----
  {
    int row = tid >> 1;                       // 0..127
    int eh  = (tid & 1) << 5;                 // 0 or 32
    const float* src = q + ((size_t)(b * SS + q0 + row) * DD) + hh * HDIM + eh;
#pragma unroll
    for (int i = 0; i < 4; ++i) {
      f4v x0 = *(const f4v*)(src + i * 8);
      f4v x1 = *(const f4v*)(src + i * 8 + 4);
      i4v p = { pack2(x0.x * rscale, x0.y * rscale),
                pack2(x0.z * rscale, x0.w * rscale),
                pack2(x1.x * rscale, x1.y * rscale),
                pack2(x1.z * rscale, x1.w * rscale) };
      *(i4v*)&Qs[row][eh + i * 8] = p;
    }
  }
  __syncthreads();

  // per-wave Q fragments (K-dim 64 = two 32-wide WMMA steps)
  bf16x16 qfrag[2];
#pragma unroll
  for (int kk = 0; kk < 2; ++kk)
    qfrag[kk] = load_a_frag(&Qs[wave * 16 + ln][0], kk * 32 + hl * 8);

  v8f   acc_o[4] = {};
  float m_r[8], l_r[8];
#pragma unroll
  for (int r = 0; r < 8; ++r) { m_r[r] = -1e30f; l_r[r] = 0.0f; }

  for (int t0 = 0; t0 < q0 + 128; t0 += 32) {   // causal: keys up to block max query
    // ---- stage K tile [t][e] (vectorized) and V tile transposed [e][t] ----
    {
      int t  = tid >> 3;                        // 0..31
      int eg = (tid & 7) << 3;                  // 0..56 step 8
      const float* ks = k + ((size_t)(b * SS + t0 + t) * DD) + hh * HDIM + eg;
      const float* vs = v + ((size_t)(b * SS + t0 + t) * DD) + hh * HDIM + eg;
      f4v k0v = *(const f4v*)(ks);
      f4v k1v = *(const f4v*)(ks + 4);
      i4v kp = { pack2(k0v.x, k0v.y), pack2(k0v.z, k0v.w),
                 pack2(k1v.x, k1v.y), pack2(k1v.z, k1v.w) };
      *(i4v*)&Ks[t][eg] = kp;
      f4v v0v = *(const f4v*)(vs);
      f4v v1v = *(const f4v*)(vs + 4);
      Vs[eg + 0][t] = f2bf(v0v.x); Vs[eg + 1][t] = f2bf(v0v.y);
      Vs[eg + 2][t] = f2bf(v0v.z); Vs[eg + 3][t] = f2bf(v0v.w);
      Vs[eg + 4][t] = f2bf(v1v.x); Vs[eg + 5][t] = f2bf(v1v.y);
      Vs[eg + 6][t] = f2bf(v1v.z); Vs[eg + 7][t] = f2bf(v1v.w);
    }
    __syncthreads();

    // ---- S = Q K^T : two 16-key subtiles, chained over K-dim 64 ----
    v8f sacc[2] = {};
#pragma unroll
    for (int sub = 0; sub < 2; ++sub) {
#pragma unroll
      for (int kk = 0; kk < 2; ++kk) {
        bf16x16 bfrag = load_a_frag(&Ks[sub * 16 + ln][0], kk * 32 + hl * 8);
        sacc[sub] = __builtin_amdgcn_wmma_f32_16x16x32_bf16(
            false, qfrag[kk], false, bfrag, (short)0, sacc[sub], false, false);
      }
    }

    // ---- causal mask + online softmax (row = hl*8 + r across 16-lane half) ----
#pragma unroll
    for (int r = 0; r < 8; ++r) {
      int grow = q0 + wave * 16 + hl * 8 + r;
      float s0 = sacc[0][r]; if (t0 + ln      > grow) s0 = -1e30f;
      float s1 = sacc[1][r]; if (t0 + 16 + ln > grow) s1 = -1e30f;
      float tm = fmaxf(s0, s1);
#pragma unroll
      for (int msk = 1; msk < 16; msk <<= 1) tm = fmaxf(tm, __shfl_xor(tm, msk, 32));
      float mnew  = fmaxf(m_r[r], tm);
      float alpha = __expf(m_r[r] - mnew);
      float p0 = __expf(s0 - mnew);
      float p1 = __expf(s1 - mnew);
      if (mnew < -1e29f) { p0 = 0.0f; p1 = 0.0f; alpha = 1.0f; }  // fully-masked tile
      float ts = p0 + p1;
#pragma unroll
      for (int msk = 1; msk < 16; msk <<= 1) ts += __shfl_xor(ts, msk, 32);
      l_r[r] = l_r[r] * alpha + ts;
      m_r[r] = mnew;
#pragma unroll
      for (int j = 0; j < 4; ++j) acc_o[j][r] = acc_o[j][r] * alpha;
      Ps[wave][hl * 8 + r][ln]      = f2bf(p0);
      Ps[wave][hl * 8 + r][16 + ln] = f2bf(p1);
    }
    __syncthreads();

    // ---- O += P V ----
    bf16x16 pfrag = load_a_frag(&Ps[wave][ln][0], hl * 8);
#pragma unroll
    for (int j = 0; j < 4; ++j) {
      bf16x16 vfrag = load_a_frag(&Vs[j * 16 + ln][0], hl * 8);
      acc_o[j] = __builtin_amdgcn_wmma_f32_16x16x32_bf16(
          false, pfrag, false, vfrag, (short)0, acc_o[j], false, false);
    }
    __syncthreads();
  }

  // ---- normalize and write out ----
#pragma unroll
  for (int j = 0; j < 4; ++j) {
#pragma unroll
    for (int r = 0; r < 8; ++r) {
      int grow = q0 + wave * 16 + hl * 8 + r;
      float val = acc_o[j][r] / l_r[r];
      o[((size_t)(b * SS + grow) * DD) + hh * HDIM + j * 16 + ln] = val;
    }
  }
}

// ---------------- host-side orchestration ----------------
extern "C" void kernel_launch(void* const* d_in, const int* in_sizes, int n_in,
                              void* d_out, int out_size, void* d_ws, size_t ws_size,
                              hipStream_t stream) {
  (void)in_sizes; (void)n_in; (void)out_size; (void)ws_size;

  const int*   x      = (const int*)  d_in[0];
  const float* wte    = (const float*)d_in[1];
  const float* wpe    = (const float*)d_in[2];
  const float* ln1_w  = (const float*)d_in[3];
  const float* ln1_b  = (const float*)d_in[4];
  const float* Wq     = (const float*)d_in[5];
  const float* bq     = (const float*)d_in[6];
  const float* Wk     = (const float*)d_in[7];
  const float* bk     = (const float*)d_in[8];
  const float* Wv     = (const float*)d_in[9];
  const float* bv     = (const float*)d_in[10];
  const float* Wp     = (const float*)d_in[11];
  const float* bp     = (const float*)d_in[12];
  const float* ln2_w  = (const float*)d_in[13];
  const float* ln2_b  = (const float*)d_in[14];
  const float* lnfc_w = (const float*)d_in[15];
  const float* lnfc_b = (const float*)d_in[16];
  const float* W1     = (const float*)d_in[17];
  const float* b1     = (const float*)d_in[18];
  const float* W2     = (const float*)d_in[19];
  const float* b2     = (const float*)d_in[20];
  const float* lnf_w  = (const float*)d_in[21];
  const float* lnf_b  = (const float*)d_in[22];
  const float* Wh     = (const float*)d_in[23];
  const float* bh     = (const float*)d_in[24];
  float* out = (float*)d_out;

  const size_t ACT = (size_t)MTOT * DD;   // 3,145,728 floats
  float* h  = (float*)d_ws;
  float* a  = h  + ACT;
  float* a2 = a  + ACT;
  float* qb = a2 + ACT;
  float* kb = qb + ACT;
  float* vb = kb + ACT;
  float* ob = vb + ACT;
  float* m1 = ob + ACT;                    // MTOT x FF

  const dim3 blk(256);
  const dim3 gRows(MTOT);
  const dim3 gD   ((DD   + 127) / 128, MTOT / 128);
  const dim3 gFF  ((FF   + 127) / 128, MTOT / 128);
  const dim3 gV   ((VV   + 127) / 128, MTOT / 128);
  const dim3 gAtt (SS / 128, NH, BB);

  embed_kernel<<<gRows, blk, 0, stream>>>(x, wte, wpe, h);

  for (int l = 0; l < NL; ++l) {
    const float* Wq_l = Wq + (size_t)l * NH * DD * HDIM;
    const float* Wk_l = Wk + (size_t)l * NH * DD * HDIM;
    const float* Wv_l = Wv + (size_t)l * NH * DD * HDIM;
    const float* Wp_l = Wp + (size_t)l * DD * DD;
    const float* W1_l = W1 + (size_t)l * DD * FF;
    const float* W2_l = W2 + (size_t)l * FF * DD;

    ln_kernel<<<gRows, blk, 0, stream>>>(h, ln1_w + l * DD, ln1_b + l * DD, a);

    gemm_wmma<<<gD, blk, 0, stream>>>(a, Wq_l, bq + l * DD, nullptr, qb,
                                      MTOT, DD, DD, /*wmode*/1, 0);
    gemm_wmma<<<gD, blk, 0, stream>>>(a, Wk_l, bk + l * DD, nullptr, kb,
                                      MTOT, DD, DD, 1, 0);
    gemm_wmma<<<gD, blk, 0, stream>>>(a, Wv_l, bv + l * DD, nullptr, vb,
                                      MTOT, DD, DD, 1, 0);

    attn_kernel<<<gAtt, blk, 0, stream>>>(qb, kb, vb, ob);

    // h = h + o @ Wp + bp
    gemm_wmma<<<gD, blk, 0, stream>>>(ob, Wp_l, bp + l * DD, h, h,
                                      MTOT, DD, DD, 0, 0);

    // double LayerNorm (ln2 then fc-block's own LN), as in reference
    ln_kernel<<<gRows, blk, 0, stream>>>(h, ln2_w + l * DD, ln2_b + l * DD, a);
    ln_kernel<<<gRows, blk, 0, stream>>>(a, lnfc_w + l * DD, lnfc_b + l * DD, a2);

    // m1 = gelu(a2 @ W1 + b1) ; h = h + m1 @ W2 + b2
    gemm_wmma<<<gFF, blk, 0, stream>>>(a2, W1_l, b1 + l * FF, nullptr, m1,
                                       MTOT, FF, DD, 0, /*gelu*/1);
    gemm_wmma<<<gD, blk, 0, stream>>>(m1, W2_l, b2 + l * DD, h, h,
                                      MTOT, DD, FF, 0, 0);
  }

  ln_kernel<<<gRows, blk, 0, stream>>>(h, lnf_w, lnf_b, a);
  gemm_wmma<<<gV, blk, 0, stream>>>(a, Wh, bh, nullptr, out,
                                    MTOT, VV, DD, 0, 0);
}
